// QKVAttention_84808424227131
// MI455X (gfx1250) — matrix operand order repaired and checked by reading, not compiled, not run on hardware
//
#include <hip/hip_runtime.h>
#include <hip/hip_bf16.h>
#include <stdint.h>

typedef _Float16 v16h __attribute__((ext_vector_type(16)));
typedef _Float16 v8h  __attribute__((ext_vector_type(8)));
typedef float    v8f  __attribute__((ext_vector_type(8)));
typedef unsigned int v4u __attribute__((ext_vector_type(4)));
typedef int          v8i __attribute__((ext_vector_type(8)));
typedef int          v4i __attribute__((ext_vector_type(4)));
typedef uint32_t u32;
typedef uint64_t u64;

#define CH_    64
#define T_LEN  2048
#define S_LEN  77
#define L_LEN  (S_LEN + T_LEN)        // 2125
#define KTILE  32
#define NTILES ((L_LEN + KTILE - 1) / KTILE)   // 67
#define LP     (NTILES * KTILE)       // 2144 (padded key length)
#define NEG_INF (-3.0e38f)

union H16 { v16h v; v8h p[2]; };

// ---------------------------------------------------------------------------
// TDM: issue a 2-D f16 tile load (tile1 rows x tile0 elems) global -> LDS.
// Descriptor bit layout per CDNA5 ISA §8.3/8.4 (D# groups 0/1; groups 2/3 zero
// for 2-D tensors). Tile is written contiguously into LDS row-major.
// Toolchain is the 6-arg clang-23 form: (v4u, v8i, v4i, v4i, v8i, i32 cpol).
// ---------------------------------------------------------------------------
__device__ __forceinline__ void tdm_load_2d_f16(u32 lds_off, const _Float16* gptr,
                                                u32 dim0, u32 dim1,
                                                u32 tile0, u32 tile1, u32 stride0)
{
    u64 ga = (u64)(uintptr_t)gptr;
    v4u g0;
    g0[0] = 1u;                                        // count=1, user D#, no gather
    g0[1] = lds_off;                                   // lds_addr (bytes)
    g0[2] = (u32)(ga & 0xffffffffu);                   // global_addr[31:0]
    g0[3] = (u32)((ga >> 32) & 0x01ffffffu) | 0x80000000u; // addr[56:32] | type=2
    v8i g1;
    g1[0] = (int)(1u << 16);                           // data_size=1 -> 2 bytes
    g1[1] = (int)(dim0 << 16);                         // tensor_dim0[15:0]
    g1[2] = (int)((dim0 >> 16) | (dim1 << 16));        // dim0[31:16] | dim1[15:0]
    g1[3] = (int)((dim1 >> 16) | (tile0 << 16));       // dim1[31:16] | tile_dim0
    g1[4] = (int)(tile1 & 0xffffu);                    // tile_dim1 (tile_dim2=0)
    g1[5] = (int)stride0;                              // tensor_dim0_stride[31:0]
    g1[6] = 0;
    g1[7] = 0;
    v4i gz4  = {0, 0, 0, 0};
    v8i gz8  = {0, 0, 0, 0, 0, 0, 0, 0};
    __builtin_amdgcn_tensor_load_to_lds(g0, g1, gz4, gz4, gz8, 0);
}

// ---------------------------------------------------------------------------
// Prep: build padded f16 K-hat [n][s][64] (scaled, concat(ek,k)) and
// V-hat [n][c][LP] (concat(ev,v)) in workspace. Pad keys [L, LP) are zero.
// ---------------------------------------------------------------------------
__global__ __launch_bounds__(256)
void prep_kv(const float* __restrict__ qkv, const float* __restrict__ ekv,
             _Float16* __restrict__ Kw, _Float16* __restrict__ Vw, int N)
{
    const float scale = 0.3535533905932738f;   // 64^(-1/4)
    int idx = blockIdx.x * 256 + threadIdx.x;
    const int per_n = LP * CH_;
    if (idx >= N * per_n) return;
    int n = idx / per_n, r = idx - n * per_n;
    {   // K-hat: [n][s][c], coalesced in c
        int s = r >> 6, c = r & 63;
        float v = 0.0f;
        if (s < S_LEN)      v = ekv[((size_t)(n * 128 + c)) * S_LEN + s];
        else if (s < L_LEN) v = qkv[((size_t)(n * 192 + 64 + c)) * T_LEN + (s - S_LEN)];
        Kw[idx] = (_Float16)(v * scale);
    }
    {   // V-hat: [n][c][s], coalesced in s
        int c = r / LP, s = r - c * LP;
        float v = 0.0f;
        if (s < S_LEN)      v = ekv[((size_t)(n * 128 + 64 + c)) * S_LEN + s];
        else if (s < L_LEN) v = qkv[((size_t)(n * 192 + 128 + c)) * T_LEN + (s - S_LEN)];
        Vw[idx] = (_Float16)v;
    }
}

// ---------------------------------------------------------------------------
// Main: 4 wave32 per block; block = (batch-head n, 64-query tile); wave = 16
// queries. Flash attention over LP keys in 32-key tiles; K/V tiles streamed
// into double-buffered LDS by the Tensor Data Mover, overlapped with WMMA.
// ---------------------------------------------------------------------------
__global__ __launch_bounds__(128)
void qkv_attn_kernel(const _Float16* __restrict__ Kw,   // [N][LP][64]  scaled f16
                     const _Float16* __restrict__ Vw,   // [N][64][LP]  f16
                     const float*    __restrict__ qkv,  // [N][192][T]
                     const float*    __restrict__ mask, // [1][T][L]
                     float*          __restrict__ out)  // [N][64][T]
{
    __shared__ _Float16 Kt[2][KTILE * CH_];   // [key][ch]
    __shared__ _Float16 Vt[2][CH_ * KTILE];   // [ch][key]

    const int tid  = threadIdx.x;
    const int lane = tid & 31;
    const int wave = tid >> 5;
    const int n    = blockIdx.y;
    const int h    = lane >> 4;      // lane half: selects K-slice of A operands
    const int ln   = lane & 15;

    const float scale = 0.3535533905932738f;

    // ---- Q B-fragments (B 32x16 f16, contiguous k order used here: lanes0-15 K=0..15, lanes16-31 K=16..31)
    const int tq = blockIdx.x * 64 + wave * 16 + ln;   // query column for this lane
    v16h qb[2];
    #pragma unroll
    for (int ch = 0; ch < 2; ++ch) {
        #pragma unroll
        for (int i = 0; i < 16; ++i) {
            int c = ch * 32 + h * 16 + i;
            qb[ch][i] = (_Float16)(qkv[((size_t)(n * 192 + c)) * T_LEN + tq] * scale);
        }
    }

    v8f   oacc[4] = {};
    float m_run = NEG_INF, l_run = 0.0f;

    // ---- prologue: TDM tile 0 into buffer 0 --------------------------------
    if (tid < 32) {
        tdm_load_2d_f16((u32)(uintptr_t)&Kt[0][0], Kw + (size_t)n * LP * CH_,
                        CH_, LP, CH_, KTILE, CH_);
        tdm_load_2d_f16((u32)(uintptr_t)&Vt[0][0], Vw + (size_t)n * CH_ * LP,
                        LP, CH_, KTILE, CH_, LP);
        __builtin_amdgcn_s_wait_tensorcnt(0);
    }
    __syncthreads();

    for (int kt = 0; kt < NTILES; ++kt) {
        const int s0  = kt * KTILE;
        const int cur = kt & 1;

        // ---- TDM prefetch of next tile into other buffer (overlaps compute)
        if (tid < 32 && kt + 1 < NTILES) {
            const int s1 = s0 + KTILE;
            tdm_load_2d_f16((u32)(uintptr_t)&Kt[cur ^ 1][0],
                            Kw + ((size_t)n * LP + s1) * CH_,
                            CH_, LP, CH_, KTILE, CH_);
            tdm_load_2d_f16((u32)(uintptr_t)&Vt[cur ^ 1][0],
                            Vw + (size_t)n * CH_ * LP + s1,
                            LP, CH_, KTILE, CH_, LP);
        }

        // ---- K A-fragments (A 16x32 f16, contiguous k order matching Q: lane<16 K{0..15}, else K{16..31})
        H16 ka[2][2];
        #pragma unroll
        for (int sub = 0; sub < 2; ++sub) {
            int srow = sub * 16 + ln;
            #pragma unroll
            for (int ch = 0; ch < 2; ++ch) {
                ka[sub][ch].p[0] = *(const v8h*)&Kt[cur][srow * CH_ + ch * 32 + h * 16];
                ka[sub][ch].p[1] = *(const v8h*)&Kt[cur][srow * CH_ + ch * 32 + h * 16 + 8];
            }
        }
        // ---- V A-fragments: rows = channels, K = 32 keys
        H16 va[4];
        #pragma unroll
        for (int cm = 0; cm < 4; ++cm) {
            int crow = cm * 16 + ln;
            va[cm].p[0] = *(const v8h*)&Vt[cur][crow * KTILE + h * 16];
            va[cm].p[1] = *(const v8h*)&Vt[cur][crow * KTILE + h * 16 + 8];
        }

        // ---- scores S^T = K_s * Q_s^T : two 16key x 16q tiles
        v8f sacc[2];
        #pragma unroll
        for (int sub = 0; sub < 2; ++sub) {
            v8f z = {};
            z = __builtin_amdgcn_wmma_f32_16x16x32_f16(false, ka[sub][0].v, false, qb[0], (short)0, z, false, false);
            z = __builtin_amdgcn_wmma_f32_16x16x32_f16(false, ka[sub][1].v, false, qb[1], (short)0, z, false, false);
            sacc[sub] = z;
        }

        // ---- online softmax (query = column = lane; keys in regs + xor-16 partner)
        float p[2][8];
        float tmax = NEG_INF;
        #pragma unroll
        for (int sub = 0; sub < 2; ++sub) {
            #pragma unroll
            for (int r = 0; r < 8; ++r) {
                int   sg = s0 + sub * 16 + h * 8 + r;
                float v  = sacc[sub][r];
                if (sg < L_LEN) v += mask[(size_t)tq * L_LEN + sg];
                else            v  = NEG_INF;            // padded keys
                p[sub][r] = v;
                tmax = fmaxf(tmax, v);
            }
        }
        tmax = fmaxf(tmax, __shfl_xor(tmax, 16, 32));
        float m_new = fmaxf(m_run, tmax);
        float corr  = __expf(m_run - m_new);
        l_run *= corr;
        #pragma unroll
        for (int cm = 0; cm < 4; ++cm)
            #pragma unroll
            for (int r = 0; r < 8; ++r) oacc[cm][r] *= corr;

        float tsum = 0.0f;
        #pragma unroll
        for (int sub = 0; sub < 2; ++sub)
            #pragma unroll
            for (int r = 0; r < 8; ++r) {
                float e = __expf(p[sub][r] - m_new);
                p[sub][r] = e;
                tsum += e;
            }
        tsum += __shfl_xor(tsum, 16, 32);
        l_run += tsum;
        m_run  = m_new;

        // ---- P (D layout) -> B operand (32key x 16q): one xor-16 exchange
        H16 pb;
        #pragma unroll
        for (int r = 0; r < 8; ++r) {
            float o0 = __shfl_xor(p[0][r], 16, 32);
            float o1 = __shfl_xor(p[1][r], 16, 32);
            pb.v[r]     = (_Float16)(h ? o1 : p[0][r]);   // key = h*16 + r
            pb.v[8 + r] = (_Float16)(h ? p[1][r] : o0);   // key = h*16 + 8 + r
        }

        // ---- PV: out[16ch x 16q] += V[16ch x 32key] * P[32key x 16q]
        #pragma unroll
        for (int cm = 0; cm < 4; ++cm)
            oacc[cm] = __builtin_amdgcn_wmma_f32_16x16x32_f16(false, va[cm].v, false, pb.v, (short)0, oacc[cm], false, false);

        // ---- close the pipeline stage: next tile resident, all waves synced
        if (tid < 32) __builtin_amdgcn_s_wait_tensorcnt(0);
        __syncthreads();
    }

    // ---- epilogue ----------------------------------------------------------
    float inv = 1.0f / l_run;
    #pragma unroll
    for (int cm = 0; cm < 4; ++cm)
        #pragma unroll
        for (int r = 0; r < 8; ++r) {
            int c = cm * 16 + h * 8 + r;
            out[((size_t)(n * CH_ + c)) * T_LEN + tq] = oacc[cm][r] * inv;
        }
}

extern "C" void kernel_launch(void* const* d_in, const int* in_sizes, int n_in,
                              void* d_out, int out_size, void* d_ws, size_t ws_size,
                              hipStream_t stream) {
    (void)n_in; (void)ws_size; (void)out_size;
    const float* qkv  = (const float*)d_in[0];   // [N, 3*64, 2048]
    const float* ekv  = (const float*)d_in[1];   // [N, 2*64, 77]
    const float* mask = (const float*)d_in[2];   // [1, 2048, 2125]
    float* out = (float*)d_out;                  // [N, 64, 2048]

    const int N = in_sizes[0] / (3 * CH_ * T_LEN);   // 32

    _Float16* Kw = (_Float16*)d_ws;                       // N*LP*64 halfs (~8.8 MB)
    _Float16* Vw = Kw + (size_t)N * LP * CH_;             // N*64*LP halfs (~8.8 MB)

    const int prep_elems = N * LP * CH_;
    prep_kv<<<(prep_elems + 255) / 256, 256, 0, stream>>>(qkv, ekv, Kw, Vw, N);

    dim3 grid(T_LEN / 64, N);
    qkv_attn_kernel<<<grid, dim3(128), 0, stream>>>(Kw, Vw, qkv, mask, out);
}
